// Attention_50105088475728
// MI455X (gfx1250) — matrix operand order repaired and checked by reading, not compiled, not run on hardware
//
#include <hip/hip_runtime.h>

#define DEVI __device__ __forceinline__

typedef __attribute__((ext_vector_type(16))) __bf16 v16bf;
typedef __attribute__((ext_vector_type(8)))  __bf16 v8bf;
typedef __attribute__((ext_vector_type(8)))  float  v8f;

constexpr int Bb  = 4;
constexpr int Ss  = 2048;
constexpr int DIN = 512;
constexpr int Hh  = 8;
constexpr int DH  = 64;
constexpr int BS  = Bb * Ss;   // 8192
constexpr int HD  = Hh * DH;   // 512

DEVI __bf16 f2bf(float x) { return (__bf16)x; }

DEVI v8bf ld8(const __bf16* p) { return *(const v8bf*)p; }

DEVI v16bf cat16(v8bf lo, v8bf hi) {
  return __builtin_shufflevector(lo, hi, 0,1,2,3,4,5,6,7,8,9,10,11,12,13,14,15);
}

// A-matrix 16x32 bf16 fragment (ISA 7.12.2): lane m=lane&15 holds row m.
// lanes 0-15: K = 0..7 (halves 0-7), 16..23 (halves 8-15); lanes 16-31: 8..15, 24..31
DEVI v16bf load_a_frag(const __bf16* tile, int ld, int lane) {
  const int m = lane & 15, hi = lane >> 4;
  const __bf16* p = tile + (size_t)m * ld + hi * 8;
  return cat16(ld8(p), ld8(p + 16));
}

// B-matrix 32x16 bf16 fragment: lane n=lane&15 holds column n;
// lanes 0-15: K = 0..7, 16..23; lanes 16-31: 8..15, 24..31. tileT is [n][k].
DEVI v16bf load_b_frag(const __bf16* tileT, size_t ld, int lane) {
  const int n = lane & 15, hi = lane >> 4;
  const __bf16* p = tileT + (size_t)n * ld + hi * 8;
  return cat16(ld8(p), ld8(p + 16));
}

DEVI v8f wmma_bf16(v16bf a, v16bf b, v8f c) {
  return __builtin_amdgcn_wmma_f32_16x16x32_bf16(
      false, a, false, b, (short)0, c, false, false);
}

// --- gfx1250 async global->LDS copy (ASYNCcnt-tracked), 16 bytes per lane ---
DEVI void async_copy16(const __bf16* g, const __bf16* lds) {
  const unsigned loff = (unsigned)(size_t)(const void*)lds;  // low 32b = LDS offset
  const unsigned long long ga = (unsigned long long)(size_t)(const void*)g;
  asm volatile("global_load_async_to_lds_b128 %0, %1, off"
               :: "v"(loff), "v"(ga) : "memory");
}

DEVI void wait_async0() {
#if __has_builtin(__builtin_amdgcn_s_wait_asynccnt)
  __builtin_amdgcn_s_wait_asynccnt(0);
#else
  asm volatile("s_wait_asynccnt 0x0" ::: "memory");
#endif
}

// ---------------------------------------------------------------------------
// Kernel 1: Y = X @ W + bias  (M=8192, N=512, K=512), f32 in, bf16 out.
// vtrans=0 -> [B*H][S][64], vtrans=1 -> [B*H][64][S].
// 256 thr (8 waves, 4M x 2N), tile 128x128, K-step 32,
// register-prefetch pipelining: next tile's global loads overlap WMMAs.
// ---------------------------------------------------------------------------
__global__ __launch_bounds__(256)
void qkv_proj_kernel(const float* __restrict__ X, const float* __restrict__ W,
                     const float* __restrict__ bias, __bf16* __restrict__ Y,
                     int vtrans) {
  __shared__ __bf16 sA[128 * 40];   // [row][k], pad 40
  __shared__ __bf16 sB[128 * 40];   // [n][k]  (transposed W tile), pad 40

  const int tid  = threadIdx.x;
  const int lane = tid & 31, wave = tid >> 5;
  const int wm = wave & 3, wn = wave >> 2;
  const int bm = blockIdx.x, bn = blockIdx.y;

  v8f c[2][4];
#pragma unroll
  for (int mi = 0; mi < 2; ++mi)
#pragma unroll
    for (int ni = 0; ni < 4; ++ni) c[mi][ni] = {};

  float4 ra[4], rb[4];
  auto load_tiles = [&](int k0) {
#pragma unroll
    for (int j = 0; j < 4; ++j) {
      const int idx = tid + j * 256;                 // 0..1023
      const int row = idx >> 3, c4 = idx & 7;       // A: 128 x (8 float4)
      ra[j] = *(const float4*)(X + (size_t)(bm * 128 + row) * DIN + k0 + c4 * 4);
      const int kk = idx >> 5, n4 = idx & 31;       // B: 32 x (32 float4)
      rb[j] = *(const float4*)(W + (size_t)(k0 + kk) * HD + bn * 128 + n4 * 4);
    }
  };
  auto store_tiles = [&]() {
#pragma unroll
    for (int j = 0; j < 4; ++j) {
      const int idx = tid + j * 256;
      const int row = idx >> 3, c4 = idx & 7;
      __bf16* d = &sA[row * 40 + c4 * 4];
      d[0] = f2bf(ra[j].x); d[1] = f2bf(ra[j].y);
      d[2] = f2bf(ra[j].z); d[3] = f2bf(ra[j].w);
      const int kk = idx >> 5, n4 = idx & 31;
      sB[(n4 * 4 + 0) * 40 + kk] = f2bf(rb[j].x);
      sB[(n4 * 4 + 1) * 40 + kk] = f2bf(rb[j].y);
      sB[(n4 * 4 + 2) * 40 + kk] = f2bf(rb[j].z);
      sB[(n4 * 4 + 3) * 40 + kk] = f2bf(rb[j].w);
    }
  };

  load_tiles(0);
  for (int k0 = 0; k0 < DIN; k0 += 32) {
    store_tiles();
    __syncthreads();
    if (k0 + 32 < DIN) load_tiles(k0 + 32);   // overlaps with WMMAs below

    const v16bf a0 = load_a_frag(&sA[(wm * 32 + 0)  * 40], 40, lane);
    const v16bf a1 = load_a_frag(&sA[(wm * 32 + 16) * 40], 40, lane);
#pragma unroll
    for (int ni = 0; ni < 4; ++ni) {
      const v16bf bf = load_b_frag(&sB[(wn * 64 + ni * 16) * 40], 40, lane);
      c[0][ni] = wmma_bf16(a0, bf, c[0][ni]);
      c[1][ni] = wmma_bf16(a1, bf, c[1][ni]);
    }
    __syncthreads();
  }

  const int hi = lane >> 4, nl = lane & 15;
#pragma unroll
  for (int mi = 0; mi < 2; ++mi)
#pragma unroll
    for (int ni = 0; ni < 4; ++ni) {
      const int n = bn * 128 + wn * 64 + ni * 16 + nl;
      const float bv = bias[n];
      const int head = n >> 6, dd = n & 63;
#pragma unroll
      for (int r = 0; r < 8; ++r) {
        const int m = bm * 128 + wm * 32 + mi * 16 + r + 8 * hi;
        const int bidx = m >> 11, s = m & (Ss - 1);
        const float val = c[mi][ni][r] + bv;
        const size_t off = vtrans
            ? (((size_t)bidx * Hh + head) * DH + dd) * Ss + s
            : (((size_t)bidx * Hh + head) * Ss + s) * DH + dd;
        Y[off] = f2bf(val);
      }
    }
}

// ---------------------------------------------------------------------------
// Kernel 2: flash attention per (b,h). Q,K: [B*H][S][64] bf16, Vt: [B*H][64][S].
// 128 thr = 4 waves; wave owns 16 query rows. 32-key tiles of K/V are staged
// in LDS *shared by all 4 waves* via async global->LDS copies, double-buffered
// so the copy of tile i+1 overlaps compute on tile i (ASYNCcnt + barrier).
// ---------------------------------------------------------------------------
__global__ __launch_bounds__(128)
void flash_attn_kernel(const __bf16* __restrict__ Q, const __bf16* __restrict__ K,
                       const __bf16* __restrict__ Vt, __bf16* __restrict__ Oa) {
  __shared__ __bf16 sK[2][32 * 72];   // [key][d], ld=72 (bank spread)
  __shared__ __bf16 sV[2][64 * 40];   // [d][key], ld=40
  __shared__ __bf16 sP[4][16 * 40];   // per-wave P bounce tile

  const int tid = threadIdx.x, lane = tid & 31, wave = tid >> 5;
  const int bh = blockIdx.y, b = bh >> 3, h = bh & 7;
  const int qBase = blockIdx.x * 64 + wave * 16;

  const __bf16* Qh = Q  + (size_t)bh * Ss * DH;
  const __bf16* Kh = K  + (size_t)bh * Ss * DH;
  const __bf16* Vh = Vt + (size_t)bh * DH * Ss;

  auto stage_kv = [&](int kt, int bb) {
    // K tile: 32 rows x 64 halves (8x16B chunks/row) = 256 chunks, 2/thread
#pragma unroll
    for (int j = 0; j < 2; ++j) {
      const int idx = tid + j * 128;
      const int row = idx >> 3, cc = idx & 7;
      async_copy16(Kh + (size_t)(kt + row) * DH + cc * 8, &sK[bb][row * 72 + cc * 8]);
    }
    // V tile: 64 rows (d) x 32 halves (4x16B chunks/row) = 256 chunks, 2/thread
#pragma unroll
    for (int j = 0; j < 2; ++j) {
      const int idx = tid + j * 128;
      const int row = idx >> 2, cc = idx & 3;
      async_copy16(Vh + (size_t)row * Ss + kt + cc * 8, &sV[bb][row * 40 + cc * 8]);
    }
  };

  // Q fragments: 16 rows x 64 head-dims = 2 A-fragments, loaded once.
  const v16bf aq0 = load_a_frag(Qh + (size_t)qBase * DH,      DH, lane);
  const v16bf aq1 = load_a_frag(Qh + (size_t)qBase * DH + 32, DH, lane);

  float mrow[8], lrow[8];
  v8f o[4];
#pragma unroll
  for (int r = 0; r < 8; ++r) { mrow[r] = -3.0e38f; lrow[r] = 0.0f; }
#pragma unroll
  for (int ni = 0; ni < 4; ++ni) o[ni] = {};

  const int hi = lane >> 4, nl = lane & 15;
  __bf16* P = sP[wave];

  stage_kv(0, 0);
  wait_async0();
  __syncthreads();

  for (int kt = 0; kt < Ss; kt += 32) {
    const int bb = (kt >> 5) & 1;
    if (kt + 32 < Ss) stage_kv(kt + 32, bb ^ 1);   // overlaps compute below

    // scores for 32 keys = two 16x16 tiles; head dim (64) = 2 K-chunks
    v8f sc[2];
#pragma unroll
    for (int t = 0; t < 2; ++t) {
      const __bf16* kt16 = &sK[bb][(t * 16) * 72];
      const v16bf bk0 = load_b_frag(kt16,      72, lane);
      const v16bf bk1 = load_b_frag(kt16 + 32, 72, lane);
      v8f acc = {};
      acc = wmma_bf16(aq0, bk0, acc);
      acc = wmma_bf16(aq1, bk1, acc);
      sc[t] = acc * 0.125f;   // 1/sqrt(64)
    }
    // online softmax; C-layout: slot r <-> query row r + 8*hi
#pragma unroll
    for (int r = 0; r < 8; ++r) {
      float mx = fmaxf(sc[0][r], sc[1][r]);
#pragma unroll
      for (int off = 1; off < 16; off <<= 1)
        mx = fmaxf(mx, __shfl_xor(mx, off, 32));
      const float mnew = fmaxf(mrow[r], mx);
      const float corr = __expf(mrow[r] - mnew);
      const float p0 = __expf(sc[0][r] - mnew);
      const float p1 = __expf(sc[1][r] - mnew);
      sc[0][r] = p0; sc[1][r] = p1;
      float rs = p0 + p1;
#pragma unroll
      for (int off = 1; off < 16; off <<= 1)
        rs += __shfl_xor(rs, off, 32);
      lrow[r] = lrow[r] * corr + rs;
      mrow[r] = mnew;
#pragma unroll
      for (int ni = 0; ni < 4; ++ni) o[ni][r] *= corr;
    }
    // C-layout -> A-layout via per-wave LDS bounce (wave-private, in-order DS)
#pragma unroll
    for (int t = 0; t < 2; ++t)
#pragma unroll
      for (int r = 0; r < 8; ++r)
        P[(r + 8 * hi) * 40 + t * 16 + nl] = f2bf(sc[t][r]);
    const v16bf ap = load_a_frag(P, 40, lane);
    // O += P @ V
#pragma unroll
    for (int ni = 0; ni < 4; ++ni) {
      const v16bf bv = load_b_frag(&sV[bb][(ni * 16) * 40], 40, lane);
      o[ni] = wmma_bf16(ap, bv, o[ni]);
    }

    wait_async0();      // next tile's async copies complete
    __syncthreads();    // visible to all waves; current buffer free next iter
  }

  // normalize and write Oa [B*S][H*DH]
#pragma unroll
  for (int r = 0; r < 8; ++r) {
    const float inv = 1.0f / lrow[r];
    const int qrow = qBase + r + 8 * hi;
#pragma unroll
    for (int ni = 0; ni < 4; ++ni) {
      const size_t off = ((size_t)b * Ss + qrow) * HD + h * DH + ni * 16 + nl;
      Oa[off] = f2bf(o[ni][r] * inv);
    }
  }
}

// ---------------------------------------------------------------------------
// Kernel 3: out = Oa @ Wo + bo  (M=8192, N=64, K=512), bf16 A, f32 out.
// A tile staged with async global->LDS copies (pure bf16 copy); B tile is
// f32->bf16 converted through registers.
// ---------------------------------------------------------------------------
__global__ __launch_bounds__(256)
void out_proj_kernel(const __bf16* __restrict__ A, const float* __restrict__ Wo,
                     const float* __restrict__ bo, float* __restrict__ out) {
  __shared__ __bf16 sA[128 * 40];
  __shared__ __bf16 sB[64 * 40];

  const int tid = threadIdx.x, lane = tid & 31, wave = tid >> 5;
  const int wm = wave & 3, wn = wave >> 2;
  const int bm = blockIdx.x;

  v8f c[2][2];
#pragma unroll
  for (int mi = 0; mi < 2; ++mi)
#pragma unroll
    for (int ni = 0; ni < 2; ++ni) c[mi][ni] = {};

  for (int k0 = 0; k0 < HD; k0 += 32) {
    // A tile: 128 rows x 32 halves = 4x16B chunks per row, async copy
#pragma unroll
    for (int j = 0; j < 2; ++j) {
      const int idx = tid + j * 256;          // 0..511
      const int row = idx >> 2, cc = idx & 3;
      async_copy16(A + (size_t)(bm * 128 + row) * HD + k0 + cc * 8,
                   &sA[row * 40 + cc * 8]);
    }
    // B tile: Wo[k][n] f32 -> sB[n][k] bf16 (overlaps with async A)
    {
      const int kk0 = tid >> 4, n40 = tid & 15;
      const float4 v0 = *(const float4*)(Wo + (size_t)(k0 + kk0) * DH + n40 * 4);
      const int idx1 = tid + 256;
      const int kk1 = idx1 >> 4, n41 = idx1 & 15;
      const float4 v1 = *(const float4*)(Wo + (size_t)(k0 + kk1) * DH + n41 * 4);
      sB[(n40 * 4 + 0) * 40 + kk0] = f2bf(v0.x);
      sB[(n40 * 4 + 1) * 40 + kk0] = f2bf(v0.y);
      sB[(n40 * 4 + 2) * 40 + kk0] = f2bf(v0.z);
      sB[(n40 * 4 + 3) * 40 + kk0] = f2bf(v0.w);
      sB[(n41 * 4 + 0) * 40 + kk1] = f2bf(v1.x);
      sB[(n41 * 4 + 1) * 40 + kk1] = f2bf(v1.y);
      sB[(n41 * 4 + 2) * 40 + kk1] = f2bf(v1.z);
      sB[(n41 * 4 + 3) * 40 + kk1] = f2bf(v1.w);
    }
    wait_async0();
    __syncthreads();

    const v16bf a0 = load_a_frag(&sA[(wm * 32 + 0)  * 40], 40, lane);
    const v16bf a1 = load_a_frag(&sA[(wm * 32 + 16) * 40], 40, lane);
#pragma unroll
    for (int ni = 0; ni < 2; ++ni) {
      const v16bf bf = load_b_frag(&sB[(wn * 32 + ni * 16) * 40], 40, lane);
      c[0][ni] = wmma_bf16(a0, bf, c[0][ni]);
      c[1][ni] = wmma_bf16(a1, bf, c[1][ni]);
    }
    __syncthreads();
  }

  const int hi = lane >> 4, nl = lane & 15;
#pragma unroll
  for (int mi = 0; mi < 2; ++mi)
#pragma unroll
    for (int ni = 0; ni < 2; ++ni) {
      const int n = wn * 32 + ni * 16 + nl;
      const float bv = bo[n];
#pragma unroll
      for (int r = 0; r < 8; ++r) {
        const int m = bm * 128 + wm * 32 + mi * 16 + r + 8 * hi;
        out[(size_t)m * DH + n] = c[mi][ni][r] + bv;
      }
    }
}

// ---------------------------------------------------------------------------
extern "C" void kernel_launch(void* const* d_in, const int* in_sizes, int n_in,
                              void* d_out, int out_size, void* d_ws, size_t ws_size,
                              hipStream_t stream) {
  (void)in_sizes; (void)n_in; (void)out_size; (void)ws_size;
  const float* q  = (const float*)d_in[0];
  const float* k  = (const float*)d_in[1];
  const float* v  = (const float*)d_in[2];
  const float* Wq = (const float*)d_in[3];
  const float* bq = (const float*)d_in[4];
  const float* Wk = (const float*)d_in[5];
  const float* bk = (const float*)d_in[6];
  const float* Wv = (const float*)d_in[7];
  const float* bv = (const float*)d_in[8];
  const float* Wo = (const float*)d_in[9];
  const float* bo = (const float*)d_in[10];
  float* out = (float*)d_out;

  char* ws = (char*)d_ws;
  const size_t seg = (size_t)BS * HD * sizeof(__bf16);   // 8 MiB each
  __bf16* Qw = (__bf16*)(ws);
  __bf16* Kw = (__bf16*)(ws + seg);
  __bf16* Vw = (__bf16*)(ws + 2 * seg);
  __bf16* Ow = (__bf16*)(ws + 3 * seg);

  const dim3 g1(BS / 128, HD / 128), b1(256);
  qkv_proj_kernel<<<g1, b1, 0, stream>>>(q, Wq, bq, Qw, 0);
  qkv_proj_kernel<<<g1, b1, 0, stream>>>(k, Wk, bk, Kw, 0);
  qkv_proj_kernel<<<g1, b1, 0, stream>>>(v, Wv, bv, Vw, 1);
  flash_attn_kernel<<<dim3(Ss / 64, Bb * Hh), dim3(128), 0, stream>>>(Qw, Kw, Vw, Ow);
  out_proj_kernel<<<dim3(BS / 128), dim3(256), 0, stream>>>(Ow, Wo, bo, out);
}
